// Swin_csp_19146964206064
// MI455X (gfx1250) — matrix operand order repaired and checked by reading, compiled, not run on hardware
//
#include <hip/hip_runtime.h>
#include <hip/hip_bf16.h>
#include <math.h>

// ---------------------------------------------------------------------------
// Swin cross-attention block on gfx1250 (MI455X), wave32 + WMMA f16->f32,
// with Tensor-Data-Mover (TDM) staging of GEMM A-tiles into LDS.
// B=16, C=256, H=W=56, WS=7, NH=8, HD=32, N=49, MLP=1024, M=B*H*W=50176.
// ---------------------------------------------------------------------------

typedef __attribute__((ext_vector_type(16))) _Float16 v16h;
typedef __attribute__((ext_vector_type(8)))  _Float16 v8h;
typedef __attribute__((ext_vector_type(8)))  float    v8f;
typedef __attribute__((ext_vector_type(4)))  unsigned int u32x4;
typedef __attribute__((ext_vector_type(4)))  int      i32x4;
typedef __attribute__((ext_vector_type(8)))  int      i32x8;

#define MTOK   50176      // B*H*W
#define CH     256
#define HW     3136       // 56*56
#define NWIN   1024       // B * 8*8 windows
#define NH_    8
#define HD_    32
#define MLP_   1024

#if defined(__AMDGCN__) && __has_builtin(__builtin_amdgcn_tensor_load_to_lds) && \
    __has_builtin(__builtin_amdgcn_s_wait_tensorcnt)
#define HAVE_TDM 1
#else
#define HAVE_TDM 0
#endif

static __device__ __forceinline__ v16h cat8(v8h lo, v8h hi) {
    return __builtin_shufflevector(lo, hi, 0,1,2,3,4,5,6,7,8,9,10,11,12,13,14,15);
}
static __device__ __forceinline__ v8f zero8f() {
    v8f z;
#pragma unroll
    for (int i = 0; i < 8; ++i) z[i] = 0.0f;
    return z;
}
static __device__ __forceinline__ v8h zero8h() {
    v8h z;
#pragma unroll
    for (int i = 0; i < 8; ++i) z[i] = (_Float16)0.0f;
    return z;
}
static __device__ __forceinline__ float wave_sum(float v) {
#pragma unroll
    for (int o = 16; o > 0; o >>= 1) v += __shfl_xor(v, o, 32);
    return v;
}

#if HAVE_TDM
// Issue a TDM load of a (rows x cols) f16 tile (row-major, contiguous in LDS)
// from global row-major matrix with row stride `stride` halves.
// D# layout per CDNA5 ISA ch.8 (group0: count/lds/global/type, group1: dims).
static __device__ __forceinline__ void tdm_load_tile_f16(
    const _Float16* gsrc, unsigned lds_off, int cols, int rows,
    int tensor_d0, int tensor_d1, int stride) {
    unsigned long long ga = (unsigned long long)(uintptr_t)gsrc;
    u32x4 g0;
    g0[0] = 1u;                                   // count=1, no gather
    g0[1] = lds_off;                              // lds_addr (bytes)
    g0[2] = (unsigned)ga;                         // global_addr[31:0]
    g0[3] = (unsigned)((ga >> 32) & 0x1FFFFFFu) | (2u << 30);  // addr hi | type=2
    i32x8 g1;
    g1[0] = 1 << 16;                              // data_size = 2 bytes
    g1[1] = (int)(((unsigned)tensor_d0 & 0xFFFFu) << 16);      // dim0 lo16
    g1[2] = (int)(((unsigned)tensor_d0 >> 16) |
                  (((unsigned)tensor_d1 & 0xFFFFu) << 16));    // dim0 hi | dim1 lo
    g1[3] = (int)(((unsigned)tensor_d1 >> 16) |
                  ((unsigned)cols << 16));                     // dim1 hi | tile_dim0
    g1[4] = rows;                                 // tile_dim1 (tile_dim2 = 0)
    g1[5] = stride;                               // tensor_dim0_stride lo32
    g1[6] = 0;                                    // stride hi | dim1_stride lo
    g1[7] = 0;
    i32x4 gz4;
#pragma unroll
    for (int i = 0; i < 4; ++i) gz4[i] = 0;
#if __clang_major__ >= 23
    i32x8 gz8;
#pragma unroll
    for (int i = 0; i < 8; ++i) gz8[i] = 0;
    __builtin_amdgcn_tensor_load_to_lds(g0, g1, gz4, gz4, gz8, 0);
#else
    __builtin_amdgcn_tensor_load_to_lds(g0, g1, gz4, gz4, 0);
#endif
}
#endif

// ---------------------------------------------------------------------------
// Weight convert:  dst[k*N + n] = (f16) src[n*K + k]   (B-matrix, K-major)
// ---------------------------------------------------------------------------
__global__ void k_wt(const float* __restrict__ src, _Float16* __restrict__ dst,
                     int N, int K) {
    int total = N * K;
    for (int i = blockIdx.x * blockDim.x + threadIdx.x; i < total;
         i += gridDim.x * blockDim.x) {
        int k = i / N, n = i - k * N;
        dst[i] = (_Float16)src[n * K + k];
    }
}

// ---------------------------------------------------------------------------
// x (B,C,H,W) f32  ->  xt (M, C) f16   (LDS tiled transpose, 32ch x 64px)
// ---------------------------------------------------------------------------
__global__ void k_transpose(const float* __restrict__ x, _Float16* __restrict__ xt) {
    __shared__ float t[32][65];
    int blk = blockIdx.x;
    int pb = blk % 49;            // 3136/64
    int cb = (blk / 49) % 8;      // 256/32
    int b  = blk / (49 * 8);
    int p0 = pb * 64, c0 = cb * 32;
    for (int i = threadIdx.x; i < 32 * 64; i += 256) {
        int ch = i >> 6, px = i & 63;
        t[ch][px] = x[((size_t)(b * CH + c0 + ch)) * HW + p0 + px];
    }
    __syncthreads();
    for (int i = threadIdx.x; i < 32 * 64; i += 256) {
        int px = i >> 5, ch = i & 31;
        xt[((size_t)(b * HW + p0 + px)) * CH + c0 + ch] = (_Float16)t[ch][px];
    }
}

// ---------------------------------------------------------------------------
// Padded attention bias:  bp[head][i*64+j], j>=49 -> -1e30, i>=49 -> 0
// ---------------------------------------------------------------------------
__global__ void k_bias(const float* __restrict__ rel_bias, float* __restrict__ bp) {
    int idx = blockIdx.x * blockDim.x + threadIdx.x;   // 8 * 4096
    if (idx >= NH_ * 4096) return;
    int head = idx >> 12;
    int ij = idx & 4095;
    int i = ij >> 6, j = ij & 63;
    float v;
    if (j >= 49) {
        v = -1e30f;
    } else if (i >= 49) {
        v = 0.0f;
    } else {
        int ry = i / 7, rx = i - ry * 7;
        int qy = j / 7, qx = j - qy * 7;
        v = rel_bias[((ry - qy + 6) * 13 + (rx - qx + 6)) * NH_ + head];
    }
    bp[idx] = v;
}

// ---------------------------------------------------------------------------
// Generic WMMA GEMM: out = A(MxK,f16) @ Bt(KxN,f16 K-major) [+bias][epilogue]
// block: 256 threads = 8 waves; tile 128(M) x 64(N); K step 32.
// A tile staged to LDS by the Tensor Data Mover (wave 0), B tile staged
// transposed by all threads for contiguous WMMA fragment reads.
// ---------------------------------------------------------------------------
enum { EPI_BIAS = 0, EPI_BIAS_SCALE = 1, EPI_GELU = 2, EPI_FINAL = 3 };

template <int EPI>
__global__ void k_gemm(const _Float16* __restrict__ A,
                       const _Float16* __restrict__ Bt,
                       const float* __restrict__ bias,
                       _Float16* __restrict__ outH,
                       float* __restrict__ outF,
                       const float* __restrict__ resid,
                       int M, int N, int K, float scale) {
    __shared__ _Float16 As[128 * 32];      // row-major [m][k]
    __shared__ _Float16 Bs[64 * 32];       // transposed [n][k]
    const int tid  = threadIdx.x;
    const int nb   = N >> 6;
    const int m0   = (blockIdx.x / nb) << 7;
    const int n0   = (blockIdx.x % nb) << 6;
    const int lane = tid & 31, wave = tid >> 5;
    const int lhi  = lane >> 4, lrow = lane & 15;

    v8f acc[4];
#pragma unroll
    for (int nt = 0; nt < 4; ++nt) acc[nt] = zero8f();

    for (int k0 = 0; k0 < K; k0 += 32) {
        // stage B tile transposed: 32x64 -> Bs[n][k]
        {
            int k = tid >> 3, no = (tid & 7) * 8;
            v8h bv = *(const v8h*)&Bt[(size_t)(k0 + k) * N + n0 + no];
#pragma unroll
            for (int i = 0; i < 8; ++i) Bs[(no + i) * 32 + k] = bv[i];
        }
#if HAVE_TDM
        // stage A tile 128x32 via Tensor Data Mover (wave 0 issues + waits)
        if (tid < 32) {
            tdm_load_tile_f16(A + (size_t)m0 * K + k0,
                              (unsigned)(uintptr_t)&As[0],
                              /*cols*/32, /*rows*/128,
                              /*tensor_d0*/K, /*tensor_d1*/M, /*stride*/K);
            __builtin_amdgcn_s_wait_tensorcnt(0);
        }
#else
#pragma unroll
        for (int j = 0; j < 2; ++j) {
            int c = tid * 2 + j;
            int row = c >> 2, co = (c & 3) * 8;
            *(v8h*)&As[row * 32 + co] =
                *(const v8h*)&A[(size_t)(m0 + row) * K + k0 + co];
        }
#endif
        if (k0 + 32 < K)  // hint next B rows into cache (global_prefetch_b8)
            __builtin_prefetch(&Bt[(size_t)(k0 + 32 + (tid >> 3)) * N + n0], 0, 0);
        __syncthreads();

        // A fragment (16x32): lane row = wave*16+lrow, 16 halves at k=16*lhi
        v16h a = *(const v16h*)&As[(wave * 16 + lrow) * 32 + 16 * lhi];
#pragma unroll
        for (int nt = 0; nt < 4; ++nt) {
            // B fragment (32x16): lane col = nt*16+lrow, 16 halves at k=16*lhi
            v16h b = *(const v16h*)&Bs[(nt * 16 + lrow) * 32 + 16 * lhi];
            acc[nt] = __builtin_amdgcn_wmma_f32_16x16x32_f16(
                false, a, false, b, (short)0, acc[nt], false, false);
        }
        __syncthreads();
    }

#pragma unroll
    for (int nt = 0; nt < 4; ++nt) {
        int gn = n0 + nt * 16 + lrow;
        float bv = bias[gn];
#pragma unroll
        for (int r = 0; r < 8; ++r) {
            int gm = m0 + wave * 16 + r + 8 * lhi;
            float v = acc[nt][r] + bv;
            if (EPI == EPI_BIAS) {
                outH[(size_t)gm * N + gn] = (_Float16)v;
            } else if (EPI == EPI_BIAS_SCALE) {
                outH[(size_t)gm * N + gn] = (_Float16)(v * scale);
            } else if (EPI == EPI_GELU) {
                float g = 0.5f * v * (1.0f + erff(v * 0.70710678118654752f));
                outH[(size_t)gm * N + gn] = (_Float16)g;
            } else {  // EPI_FINAL: +bias +residual, scatter to (B,C,H,W) f32
                float o = v + resid[(size_t)gm * CH + gn];
                int b = gm / HW, p = gm - b * HW;
                outF[((size_t)(b * CH + gn)) * HW + p] = o;
            }
        }
    }
}

// ---------------------------------------------------------------------------
// Fused double LayerNorm: t = LN1(e_row); optional store t (f32, raster);
// store LN2(t) as f16 at window-permuted row. One wave per token.
// ---------------------------------------------------------------------------
__global__ void k_double_ln(const _Float16* __restrict__ e,
                            const float* __restrict__ w1, const float* __restrict__ b1,
                            const float* __restrict__ w2, const float* __restrict__ b2,
                            float* __restrict__ yout,        // nullable
                            _Float16* __restrict__ awin) {
    int token = blockIdx.x * 8 + (threadIdx.x >> 5);
    int lane  = threadIdx.x & 31;
    float x[8];
    float s = 0.f, ss = 0.f;
#pragma unroll
    for (int j = 0; j < 8; ++j) {
        x[j] = (float)e[(size_t)token * CH + lane + 32 * j];
        s += x[j]; ss += x[j] * x[j];
    }
    s = wave_sum(s); ss = wave_sum(ss);
    float mean = s * (1.0f / CH);
    float rstd = rsqrtf(ss * (1.0f / CH) - mean * mean + 1e-5f);
    float t[8];
    float s2 = 0.f, ss2 = 0.f;
#pragma unroll
    for (int j = 0; j < 8; ++j) {
        int c = lane + 32 * j;
        t[j] = (x[j] - mean) * rstd * w1[c] + b1[c];
        if (yout) yout[(size_t)token * CH + c] = t[j];
        s2 += t[j]; ss2 += t[j] * t[j];
    }
    s2 = wave_sum(s2); ss2 = wave_sum(ss2);
    float mean2 = s2 * (1.0f / CH);
    float rstd2 = rsqrtf(ss2 * (1.0f / CH) - mean2 * mean2 + 1e-5f);
    // raster -> windowed permutation
    int b = token / HW, p = token - b * HW;
    int h = p / 56, w = p - h * 56;
    int tw = (b * 64 + (h / 7) * 8 + (w / 7)) * 49 + (h % 7) * 7 + (w % 7);
#pragma unroll
    for (int j = 0; j < 8; ++j) {
        int c = lane + 32 * j;
        awin[(size_t)tw * CH + c] = (_Float16)((t[j] - mean2) * rstd2 * w2[c] + b2[c]);
    }
}

// Single LN: f32 in (raster), f16 out (raster). One wave per token.
__global__ void k_single_ln(const float* __restrict__ y,
                            const float* __restrict__ w, const float* __restrict__ b,
                            _Float16* __restrict__ out) {
    int token = blockIdx.x * 8 + (threadIdx.x >> 5);
    int lane  = threadIdx.x & 31;
    float x[8]; float s = 0.f, ss = 0.f;
#pragma unroll
    for (int j = 0; j < 8; ++j) {
        x[j] = y[(size_t)token * CH + lane + 32 * j];
        s += x[j]; ss += x[j] * x[j];
    }
    s = wave_sum(s); ss = wave_sum(ss);
    float mean = s * (1.0f / CH);
    float rstd = rsqrtf(ss * (1.0f / CH) - mean * mean + 1e-5f);
#pragma unroll
    for (int j = 0; j < 8; ++j) {
        int c = lane + 32 * j;
        out[(size_t)token * CH + c] = (_Float16)((x[j] - mean) * rstd * w[c] + b[c]);
    }
}

// ---------------------------------------------------------------------------
// Windowed attention, one block per (window, head). 49 tokens padded to 64.
// S = Q K^T (+padded bias), softmax rows, O = P V. Block = 128 threads.
// ---------------------------------------------------------------------------
__global__ void k_attn(const _Float16* __restrict__ Q,     // (M,256) scaled q
                       const _Float16* __restrict__ KV,    // (M,512): k|v
                       const float* __restrict__ biasP,    // (8,64,64) padded
                       _Float16* __restrict__ OW) {        // (M,256)
    __shared__ _Float16 Qs[64 * 32];
    __shared__ _Float16 Ks[64 * 32];
    __shared__ _Float16 Vt[32 * 64];   // transposed: Vt[d][j]
    __shared__ float    Sf[64 * 64];
    __shared__ _Float16 Ps[64 * 64];

    const int win  = blockIdx.x >> 3;
    const int head = blockIdx.x & 7;
    const int tid  = threadIdx.x;
    const float* bh = biasP + head * 4096;

    {   // stage Q, K, V (rows >= 49 zero-padded)
        int row = tid >> 1, half = tid & 1;
        if (row < 49) {
            size_t base = (size_t)(win * 49 + row);
            const v8h* sq = (const v8h*)&Q[base * CH + head * HD_ + half * 16];
            ((v8h*)&Qs[row * 32 + half * 16])[0] = sq[0];
            ((v8h*)&Qs[row * 32 + half * 16])[1] = sq[1];
            const v8h* sk = (const v8h*)&KV[base * 512 + head * HD_ + half * 16];
            ((v8h*)&Ks[row * 32 + half * 16])[0] = sk[0];
            ((v8h*)&Ks[row * 32 + half * 16])[1] = sk[1];
            const _Float16* sv = &KV[base * 512 + 256 + head * HD_ + half * 16];
#pragma unroll
            for (int i = 0; i < 16; ++i) Vt[(half * 16 + i) * 64 + row] = sv[i];
        } else {
            v8h z = zero8h();
            ((v8h*)&Qs[row * 32 + half * 16])[0] = z;
            ((v8h*)&Qs[row * 32 + half * 16])[1] = z;
            ((v8h*)&Ks[row * 32 + half * 16])[0] = z;
            ((v8h*)&Ks[row * 32 + half * 16])[1] = z;
#pragma unroll
            for (int i = 0; i < 16; ++i) Vt[(half * 16 + i) * 64 + row] = (_Float16)0.f;
        }
    }
    __syncthreads();

    const int lane = tid & 31, wave = tid >> 5;
    const int lhi = lane >> 4, lrow = lane & 15;

    // S = Q @ K^T + bias : wave handles m-tile = wave; 4 n-tiles; K=32
    {
        v16h a = *(const v16h*)&Qs[(wave * 16 + lrow) * 32 + 16 * lhi];
#pragma unroll
        for (int nt = 0; nt < 4; ++nt) {
            v16h bfr = *(const v16h*)&Ks[(nt * 16 + lrow) * 32 + 16 * lhi];
            v8f s = __builtin_amdgcn_wmma_f32_16x16x32_f16(
                false, a, false, bfr, (short)0, zero8f(), false, false);
#pragma unroll
            for (int r = 0; r < 8; ++r) {
                int i = wave * 16 + r + 8 * lhi;   // query token
                int j = nt * 16 + lrow;            // key token
                Sf[i * 64 + j] = s[r] + bh[i * 64 + j];
            }
        }
    }
    __syncthreads();

    // softmax per row (rows 49..63 -> zero)
    if (tid < 64) {
        int i = tid;
        if (i < 49) {
            float mx = -1e30f;
            for (int j = 0; j < 64; ++j) mx = fmaxf(mx, Sf[i * 64 + j]);
            float sum = 0.f;
            for (int j = 0; j < 64; ++j) {
                float e = __expf(Sf[i * 64 + j] - mx);
                sum += e;
                Ps[i * 64 + j] = (_Float16)e;
            }
            float inv = 1.0f / sum;
            for (int j = 0; j < 64; ++j)
                Ps[i * 64 + j] = (_Float16)((float)Ps[i * 64 + j] * inv);
        } else {
            for (int j = 0; j < 64; ++j) Ps[i * 64 + j] = (_Float16)0.f;
        }
    }
    __syncthreads();

    // O = P @ V : m-tile = wave, 2 n-tiles (HD=32), K=64 (2 steps)
#pragma unroll
    for (int nt = 0; nt < 2; ++nt) {
        v8f o = zero8f();
#pragma unroll
        for (int ks = 0; ks < 2; ++ks) {
            v16h ap = *(const v16h*)&Ps[(wave * 16 + lrow) * 64 + ks * 32 + 16 * lhi];
            v16h bp = *(const v16h*)&Vt[(nt * 16 + lrow) * 64 + ks * 32 + 16 * lhi];
            o = __builtin_amdgcn_wmma_f32_16x16x32_f16(
                false, ap, false, bp, (short)0, o, false, false);
        }
#pragma unroll
        for (int r = 0; r < 8; ++r) {
            int i = wave * 16 + r + 8 * lhi;
            if (i < 49)
                OW[(size_t)(win * 49 + i) * CH + head * HD_ + nt * 16 + lrow] =
                    (_Float16)o[r];
        }
    }
}

// ---------------------------------------------------------------------------
// y_f32[raster(t), c] += po_win[t, c]   (un-window + residual add)
// ---------------------------------------------------------------------------
__global__ void k_resid(const _Float16* __restrict__ po, float* __restrict__ y) {
    int total = MTOK * CH;
    for (int idx = blockIdx.x * blockDim.x + threadIdx.x; idx < total;
         idx += gridDim.x * blockDim.x) {
        int t = idx >> 8, c = idx & 255;
        int win = t / 49, inner = t - win * 49;
        int b = win >> 6, wy = (win & 63) >> 3, wx = win & 7;
        int iy = inner / 7, ix = inner - iy * 7;
        int r = b * HW + (wy * 7 + iy) * 56 + (wx * 7 + ix);
        y[(size_t)r * CH + c] += (float)po[idx];
    }
}

// ---------------------------------------------------------------------------
extern "C" void kernel_launch(void* const* d_in, const int* in_sizes, int n_in,
                              void* d_out, int out_size, void* d_ws, size_t ws_size,
                              hipStream_t stream) {
    const float* x1       = (const float*)d_in[0];
    const float* x2       = (const float*)d_in[1];
    const float* pe_w     = (const float*)d_in[2];
    const float* pe_b     = (const float*)d_in[3];
    const float* pe_ln_w  = (const float*)d_in[4];
    const float* pe_ln_b  = (const float*)d_in[5];
    const float* n1_w     = (const float*)d_in[6];
    const float* n1_b     = (const float*)d_in[7];
    const float* q_w      = (const float*)d_in[8];
    const float* q_b      = (const float*)d_in[9];
    const float* kv_w     = (const float*)d_in[10];
    const float* kv_b     = (const float*)d_in[11];
    const float* rel_bias = (const float*)d_in[12];
    const float* proj_w   = (const float*)d_in[13];
    const float* proj_b   = (const float*)d_in[14];
    const float* n2_w     = (const float*)d_in[15];
    const float* n2_b     = (const float*)d_in[16];
    const float* fc1_w    = (const float*)d_in[17];
    const float* fc1_b    = (const float*)d_in[18];
    const float* fc2_w    = (const float*)d_in[19];
    const float* fc2_b    = (const float*)d_in[20];
    float* out = (float*)d_out;

    char* ws = (char*)d_ws;
    size_t off = 0;
    auto alloc = [&](size_t bytes) -> void* {
        off = (off + 255) & ~(size_t)255;
        void* p = ws + off;
        off += bytes;
        return p;
    };

    _Float16* Wpe = (_Float16*)alloc((size_t)CH * CH * 2);
    _Float16* Wq  = (_Float16*)alloc((size_t)CH * CH * 2);
    _Float16* Wkv = (_Float16*)alloc((size_t)CH * 512 * 2);
    _Float16* Wpj = (_Float16*)alloc((size_t)CH * CH * 2);
    _Float16* W1  = (_Float16*)alloc((size_t)CH * MLP_ * 2);
    _Float16* W2  = (_Float16*)alloc((size_t)MLP_ * CH * 2);
    float*    BP  = (float*)   alloc((size_t)NH_ * 64 * 64 * 4);
    _Float16* XT1 = (_Float16*)alloc((size_t)MTOK * CH * 2);
    _Float16* XT2 = (_Float16*)alloc((size_t)MTOK * CH * 2);
    _Float16* E1  = (_Float16*)alloc((size_t)MTOK * CH * 2);
    _Float16* E2  = (_Float16*)alloc((size_t)MTOK * CH * 2);
    float*    Y1  = (float*)   alloc((size_t)MTOK * CH * 4);
    _Float16* A1  = (_Float16*)alloc((size_t)MTOK * CH * 2);
    _Float16* A2  = (_Float16*)alloc((size_t)MTOK * CH * 2);
    _Float16* Qb  = (_Float16*)alloc((size_t)MTOK * CH * 2);
    _Float16* KVb = (_Float16*)alloc((size_t)MTOK * 512 * 2);
    _Float16* OWb = (_Float16*)alloc((size_t)MTOK * CH * 2);
    _Float16* PO  = (_Float16*)alloc((size_t)MTOK * CH * 2);
    _Float16* HM  = (_Float16*)alloc((size_t)MTOK * CH * 2);
    _Float16* HG  = (_Float16*)alloc((size_t)MTOK * MLP_ * 2);

    // 1) weight transposes (K-major f16) + padded bias
    k_wt<<<256, 256, 0, stream>>>(pe_w,  Wpe, CH,   CH);
    k_wt<<<256, 256, 0, stream>>>(q_w,   Wq,  CH,   CH);
    k_wt<<<256, 256, 0, stream>>>(kv_w,  Wkv, 512,  CH);
    k_wt<<<256, 256, 0, stream>>>(proj_w,Wpj, CH,   CH);
    k_wt<<<512, 256, 0, stream>>>(fc1_w, W1,  MLP_, CH);
    k_wt<<<512, 256, 0, stream>>>(fc2_w, W2,  CH,   MLP_);
    k_bias<<<(NH_ * 4096 + 255) / 256, 256, 0, stream>>>(rel_bias, BP);

    // 2) NCHW -> (token, C) f16
    k_transpose<<<16 * 8 * 49, 256, 0, stream>>>(x1, XT1);
    k_transpose<<<16 * 8 * 49, 256, 0, stream>>>(x2, XT2);

    const int MB = MTOK / 128;  // 392

    // 3) patch embed GEMMs
    k_gemm<EPI_BIAS><<<MB * (CH / 64), 256, 0, stream>>>(
        XT1, Wpe, pe_b, E1, nullptr, nullptr, MTOK, CH, CH, 1.0f);
    k_gemm<EPI_BIAS><<<MB * (CH / 64), 256, 0, stream>>>(
        XT2, Wpe, pe_b, E2, nullptr, nullptr, MTOK, CH, CH, 1.0f);

    // 4) fused LN(pe) -> [Y1 f32] -> LN(n1) -> windowed f16
    k_double_ln<<<MTOK / 8, 256, 0, stream>>>(E1, pe_ln_w, pe_ln_b, n1_w, n1_b, Y1, A1);
    k_double_ln<<<MTOK / 8, 256, 0, stream>>>(E2, pe_ln_w, pe_ln_b, n1_w, n1_b, nullptr, A2);

    // 5) Q (pre-scaled by HD^-0.5) and KV projections
    k_gemm<EPI_BIAS_SCALE><<<MB * (CH / 64), 256, 0, stream>>>(
        A1, Wq, q_b, Qb, nullptr, nullptr, MTOK, CH, CH, 0.17677669529663688f);
    k_gemm<EPI_BIAS><<<MB * (512 / 64), 256, 0, stream>>>(
        A2, Wkv, kv_b, KVb, nullptr, nullptr, MTOK, 512, CH, 1.0f);

    // 6) windowed attention (1024 windows x 8 heads)
    k_attn<<<NWIN * NH_, 128, 0, stream>>>(Qb, KVb, BP, OWb);

    // 7) output projection + residual (with un-window)
    k_gemm<EPI_BIAS><<<MB * (CH / 64), 256, 0, stream>>>(
        OWb, Wpj, proj_b, PO, nullptr, nullptr, MTOK, CH, CH, 1.0f);
    k_resid<<<2048, 256, 0, stream>>>(PO, Y1);

    // 8) MLP: LN(n2) -> fc1+GELU -> fc2 + residual -> (B,C,H,W) f32
    k_single_ln<<<MTOK / 8, 256, 0, stream>>>(Y1, n2_w, n2_b, HM);
    k_gemm<EPI_GELU><<<MB * (MLP_ / 64), 256, 0, stream>>>(
        HM, W1, fc1_b, HG, nullptr, nullptr, MTOK, MLP_, CH, 1.0f);
    k_gemm<EPI_FINAL><<<MB * (CH / 64), 256, 0, stream>>>(
        HG, W2, fc2_b, nullptr, out, Y1, MTOK, CH, MLP_, 1.0f);
}
